// memory_43284680409245
// MI455X (gfx1250) — compile-verified, matrix operand order checked by reading, not run on hardware
//
#include <hip/hip_runtime.h>
#include <hip/hip_bf16.h>
#include <math.h>

#define BATCH     1024
#define KDIM      128
#define MSIZE     131072
#define CHOOSEK   256
#define NBINS     512
#define TMIN      (-14.0f)
#define TRANGE    (16.0f)
#define BINW      (TRANGE / (float)NBINS)
#define INV_BINW  ((float)NBINS / TRANGE)
#define ROWS          32                     // M-tile: two 16-row A fragments
#define WAVES_PER_WG  8
#define COLS_PER_ITER (WAVES_PER_WG * 16)    // 128 memory slots / iter
#define MCHUNK        4096
#define NCHUNKS       (MSIZE / MCHUNK)       // 32
#define ITERS         (MCHUNK / COLS_PER_ITER) // 32 (even -> clean unroll-by-2)

typedef __bf16 v16bf __attribute__((ext_vector_type(16)));
typedef float  v8f   __attribute__((ext_vector_type(8)));

union FragU { v16bf v; uint4 q[2]; };
struct BTile { FragU b[4]; };                // one 16-col x 128-K key tile

// ---- fp32 -> bf16 (RNE) packed conversion ------------------------------
__global__ void prep_bf16(const float* __restrict__ src,
                          unsigned short* __restrict__ dst, int n) {
    int stride = gridDim.x * blockDim.x;
    for (int i = blockIdx.x * blockDim.x + threadIdx.x; i < n; i += stride) {
        unsigned int u = __float_as_uint(src[i]);
        unsigned int r = (u + 0x7FFFu + ((u >> 16) & 1u)) >> 16;
        dst[i] = (unsigned short)r;
    }
}

__global__ void prep_logh(const float* __restrict__ hist,
                          float* __restrict__ logh, int n) {
    int stride = gridDim.x * blockDim.x;
    for (int i = blockIdx.x * blockDim.x + threadIdx.x; i < n; i += stride)
        logh[i] = __logf(hist[i] + 1e-8f);
}

// ---- ISA 16-bit A/B fragment layout: two 16B chunks per lane -----------
__device__ __forceinline__ v16bf load_frag(const unsigned short* __restrict__ row,
                                           int kk, int half) {
    FragU f;
    const int e = kk * 32 + half * 8;
    f.q[0] = *(const uint4*)(row + e);
    f.q[1] = *(const uint4*)(row + e + 16);
    return f.v;
}

__device__ __forceinline__ void load_btile(BTile& t,
                                           const unsigned short* __restrict__ krow,
                                           int half) {
    #pragma unroll
    for (int kk = 0; kk < 4; ++kk) {
        const int e = kk * 32 + half * 8;
        t.b[kk].q[0] = *(const uint4*)(krow + e);
        t.b[kk].q[1] = *(const uint4*)(krow + e + 16);
    }
}

// 8 WMMAs on two independent accumulators (no D->A/B RAW hazards back-to-back)
__device__ __forceinline__ void wmma_2tiles(v8f& acc0, v8f& acc1,
                                            const FragU* __restrict__ a0,
                                            const FragU* __restrict__ a1,
                                            const BTile& bt) {
    #pragma unroll
    for (int kk = 0; kk < 4; ++kk) {
        acc0 = __builtin_amdgcn_wmma_f32_16x16x32_bf16(false, a0[kk].v, false, bt.b[kk].v,
                                                       (short)0, acc0, false, false);
        acc1 = __builtin_amdgcn_wmma_f32_16x16x32_bf16(false, a1[kk].v, false, bt.b[kk].v,
                                                       (short)0, acc1, false, false);
    }
}

// ---- Pass 1: WMMA GEMM tiles -> per-row histogram of t = sim + log h ---
__global__ __launch_bounds__(256) void phase1_hist(
        const unsigned short* __restrict__ qb,
        const unsigned short* __restrict__ kb,
        const float* __restrict__ logh,
        unsigned int* __restrict__ ghist) {
    __shared__ unsigned int lh[ROWS * NBINS];        // 64 KB
    const int tid  = threadIdx.x;
    const int lane = tid & 31;
    const int wave = tid >> 5;
    const int half = lane >> 4;
    const int l15  = lane & 15;
    const int r0   = blockIdx.x * ROWS;
    const int c0   = blockIdx.y * MCHUNK;

    for (int i = tid; i < ROWS * NBINS; i += blockDim.x) lh[i] = 0u;

    FragU a0f[4], a1f[4];
    {
        const unsigned short* q0 = qb + (size_t)(r0 + l15) * KDIM;
        const unsigned short* q1 = qb + (size_t)(r0 + 16 + l15) * KDIM;
        #pragma unroll
        for (int kk = 0; kk < 4; ++kk) {
            a0f[kk].v = load_frag(q0, kk, half);
            a1f[kk].v = load_frag(q1, kk, half);
        }
    }
    __syncthreads();

    const int nbase = c0 + wave * 16 + l15;
    BTile tb0, tb1;
    load_btile(tb0, kb + (size_t)nbase * KDIM, half);

    auto process = [&](int it, const BTile& bt) {
        const int n = nbase + it * COLS_PER_ITER;
        v8f acc0 = {}, acc1 = {};
        wmma_2tiles(acc0, acc1, a0f, a1f, bt);
        const float lhn = logh[n];
        #pragma unroll
        for (int i = 0; i < 8; ++i) {
            float t0 = acc0[i] + lhn;
            float t1 = acc1[i] + lhn;
            int b0 = (int)((t0 - TMIN) * INV_BINW);
            int b1 = (int)((t1 - TMIN) * INV_BINW);
            b0 = b0 < 0 ? 0 : (b0 > NBINS - 1 ? NBINS - 1 : b0);
            b1 = b1 < 0 ? 0 : (b1 > NBINS - 1 ? NBINS - 1 : b1);
            atomicAdd(&lh[(half * 8 + i) * NBINS + b0], 1u);
            atomicAdd(&lh[(16 + half * 8 + i) * NBINS + b1], 1u);
        }
    };

    for (int it = 0; it < ITERS; it += 2) {
        load_btile(tb1, kb + (size_t)(nbase + (it + 1) * COLS_PER_ITER) * KDIM, half);
        process(it, tb0);
        const int pre = (it + 2 < ITERS) ? (it + 2) : 0;   // tail: harmless L2 re-read
        load_btile(tb0, kb + (size_t)(nbase + pre * COLS_PER_ITER) * KDIM, half);
        process(it + 1, tb1);
    }

    __syncthreads();
    for (int i = tid; i < ROWS * NBINS; i += blockDim.x) {
        unsigned int c = lh[i];
        if (c) atomicAdd(&ghist[(size_t)r0 * NBINS + i], c);
    }
}

// ---- rank-256 threshold per row ----------------------------------------
__global__ void phase2_thresh(const unsigned int* __restrict__ ghist,
                              float* __restrict__ thr) {
    int row = blockIdx.x * blockDim.x + threadIdx.x;
    if (row >= BATCH) return;
    const unsigned int* h = ghist + (size_t)row * NBINS;
    unsigned int cum = 0;
    int b = 0;
    for (int i = NBINS - 1; i >= 0; --i) {
        cum += h[i];
        if (cum >= CHOOSEK) { b = i; break; }
    }
    thr[row] = TMIN + (float)b * BINW;
}

// ---- Pass 2: recompute tiles, accumulate sum(w), sum(w*v) above thr ----
__global__ __launch_bounds__(256) void phase3_accum(
        const unsigned short* __restrict__ qb,
        const unsigned short* __restrict__ kb,
        const float* __restrict__ logh,
        const float* __restrict__ vals,
        const float* __restrict__ thr,
        float* __restrict__ gsum) {            // [BATCH][2]
    __shared__ float sthr[ROWS];
    __shared__ float sacc[2 * ROWS];
    const int tid  = threadIdx.x;
    const int lane = tid & 31;
    const int wave = tid >> 5;
    const int half = lane >> 4;
    const int l15  = lane & 15;
    const int r0   = blockIdx.x * ROWS;
    const int c0   = blockIdx.y * MCHUNK;

    if (tid < ROWS)     sthr[tid] = thr[r0 + tid];
    if (tid < 2 * ROWS) sacc[tid] = 0.0f;

    FragU a0f[4], a1f[4];
    {
        const unsigned short* q0 = qb + (size_t)(r0 + l15) * KDIM;
        const unsigned short* q1 = qb + (size_t)(r0 + 16 + l15) * KDIM;
        #pragma unroll
        for (int kk = 0; kk < 4; ++kk) {
            a0f[kk].v = load_frag(q0, kk, half);
            a1f[kk].v = load_frag(q1, kk, half);
        }
    }
    __syncthreads();

    float accw0[8]  = {0,0,0,0,0,0,0,0}, accw1[8]  = {0,0,0,0,0,0,0,0};
    float accwv0[8] = {0,0,0,0,0,0,0,0}, accwv1[8] = {0,0,0,0,0,0,0,0};

    const int nbase = c0 + wave * 16 + l15;
    BTile tb0, tb1;
    load_btile(tb0, kb + (size_t)nbase * KDIM, half);

    auto process = [&](int it, const BTile& bt) {
        const int n = nbase + it * COLS_PER_ITER;
        v8f acc0 = {}, acc1 = {};
        wmma_2tiles(acc0, acc1, a0f, a1f, bt);
        const float lhn = logh[n];
        const float vn  = vals[n];
        #pragma unroll
        for (int i = 0; i < 8; ++i) {
            const int m0 = half * 8 + i;
            const int m1 = 16 + m0;
            float t0 = acc0[i] + lhn;
            float t1 = acc1[i] + lhn;
            if (t0 >= sthr[m0]) { float w = __expf(t0); accw0[i] += w; accwv0[i] += w * vn; }
            if (t1 >= sthr[m1]) { float w = __expf(t1); accw1[i] += w; accwv1[i] += w * vn; }
        }
    };

    for (int it = 0; it < ITERS; it += 2) {
        load_btile(tb1, kb + (size_t)(nbase + (it + 1) * COLS_PER_ITER) * KDIM, half);
        process(it, tb0);
        const int pre = (it + 2 < ITERS) ? (it + 2) : 0;
        load_btile(tb0, kb + (size_t)(nbase + pre * COLS_PER_ITER) * KDIM, half);
        process(it + 1, tb1);
    }

    #pragma unroll
    for (int i = 0; i < 8; ++i) {
        const int m0 = half * 8 + i;
        const int m1 = 16 + m0;
        atomicAdd(&sacc[m0], accw0[i]);
        atomicAdd(&sacc[m1], accw1[i]);
        atomicAdd(&sacc[ROWS + m0], accwv0[i]);
        atomicAdd(&sacc[ROWS + m1], accwv1[i]);
    }
    __syncthreads();
    if (tid < ROWS) {
        atomicAdd(&gsum[(size_t)(r0 + tid) * 2 + 0], sacc[tid]);
        atomicAdd(&gsum[(size_t)(r0 + tid) * 2 + 1], sacc[ROWS + tid]);
    }
}

// ---- final: normalize + clip -------------------------------------------
__global__ void phase4_out(const float* __restrict__ gsum,
                           float* __restrict__ out) {
    int row = blockIdx.x * blockDim.x + threadIdx.x;
    if (row >= BATCH) return;
    float w  = gsum[(size_t)row * 2 + 0];
    float wv = gsum[(size_t)row * 2 + 1];
    float p  = wv / fmaxf(w, 1e-30f);
    out[row] = fminf(fmaxf(p, 1e-3f), 1.0f - 1e-3f);
}

extern "C" void kernel_launch(void* const* d_in, const int* in_sizes, int n_in,
                              void* d_out, int out_size, void* d_ws, size_t ws_size,
                              hipStream_t stream) {
    (void)in_sizes; (void)n_in; (void)out_size; (void)ws_size;
    const float* q     = (const float*)d_in[0];
    const float* mkey  = (const float*)d_in[1];
    const float* mvals = (const float*)d_in[2];
    const float* mhist = (const float*)d_in[3];

    char* ws = (char*)d_ws;
    unsigned short* kb   = (unsigned short*)(ws + 0);          // 33,554,432 B
    unsigned short* qb   = (unsigned short*)(ws + 33554432);   //    262,144 B
    float*          logh = (float*)        (ws + 33816576);    //    524,288 B
    unsigned int*  ghist = (unsigned int*) (ws + 34340864);    //  2,097,152 B
    float*          thr  = (float*)        (ws + 36438016);    //      4,096 B
    float*          gsum = (float*)        (ws + 36442112);    //      8,192 B

    hipMemsetAsync(ghist, 0, (size_t)BATCH * NBINS * sizeof(unsigned int), stream);
    hipMemsetAsync(gsum,  0, (size_t)BATCH * 2 * sizeof(float), stream);

    prep_bf16<<<4096, 256, 0, stream>>>(mkey, kb, MSIZE * KDIM);
    prep_bf16<<<512,  256, 0, stream>>>(q,    qb, BATCH * KDIM);
    prep_logh<<<512,  256, 0, stream>>>(mhist, logh, MSIZE);

    dim3 grid(BATCH / ROWS, NCHUNKS);   // 32 x 32 = 1024 workgroups
    phase1_hist <<<grid, 256, 0, stream>>>(qb, kb, logh, ghist);
    phase2_thresh<<<BATCH / 256, 256, 0, stream>>>(ghist, thr);
    phase3_accum<<<grid, 256, 0, stream>>>(qb, kb, logh, mvals, thr, gsum);
    phase4_out  <<<BATCH / 256, 256, 0, stream>>>(gsum, (float*)d_out);
}
